// VectorQuantizer_23192823399191
// MI455X (gfx1250) — compile-verified, matrix operand order checked by reading, not run on hardware
//
#include <hip/hip_runtime.h>
#include <math.h>

typedef __bf16 bf16_t;
typedef __bf16 v16bf __attribute__((ext_vector_type(16)));
typedef float  v8f   __attribute__((ext_vector_type(8)));
typedef unsigned short u16x8  __attribute__((ext_vector_type(8)));
typedef unsigned short u16x16 __attribute__((ext_vector_type(16)));
typedef int v4i_vs __attribute__((vector_size(4 * sizeof(int))));

#define NUM_CODES   8192
#define EMBED_DIM   256
#define N_VEC       32768   /* 32 * 32 * 32 */
#define HW          1024    /* 32 * 32 */
#define ROWS_PER_WG 128
#define NTILES      (NUM_CODES / 16)
#define Q_OFF       1
#define Q_SIZE      8388608 /* 32*256*32*32 */
#define PERP_OFF    (Q_OFF + Q_SIZE)
#define IDX_OFF     (PERP_OFF + 1)
#define CB_PAD      264     /* bf16 row stride: 528B = 132 dwords, 132%64=4 -> conflict-free */

#if defined(__gfx1250__) && __has_builtin(__builtin_amdgcn_global_load_async_to_lds_b128)
#define USE_ASYNC 1
#else
#define USE_ASYNC 0
#endif

union FragCast { u16x16 u; v16bf b; };

static __device__ inline v16bf combine8(u16x8 g0, u16x8 g1) {
  FragCast f;
  f.u = __builtin_shufflevector(g0, g1, 0,1,2,3,4,5,6,7,8,9,10,11,12,13,14,15);
  return f.b;
}

#if USE_ASYNC
static __device__ inline void async_cp16(const float* g, float* l) {
  // 16B per lane, memory -> LDS, tracked by ASYNCcnt.
  // Builtin signature (from clang diagnostic): param0 = int4 __device__* (as1),
  // param1 = int4 __shared__* (as3), then offset, cpol.
  __builtin_amdgcn_global_load_async_to_lds_b128(
      (__attribute__((address_space(1))) v4i_vs*)g,
      (__attribute__((address_space(3))) v4i_vs*)l, 0, 0);
}
static __device__ inline void wait_async() {
#if __has_builtin(__builtin_amdgcn_s_wait_asynccnt)
  __builtin_amdgcn_s_wait_asynccnt(0);
#else
  asm volatile("s_wait_asynccnt 0" ::: "memory");
#endif
}
#endif

// ---------------------------------------------------------------- prep:
// per-code squared norms + zero the histogram (ws is poisoned between runs)
__global__ void vq_prep(const float* __restrict__ cb,
                        float* __restrict__ cnorm,
                        unsigned* __restrict__ counts) {
  __shared__ float s[256];
  const int b = blockIdx.x;
  const int t = threadIdx.x;
  float v = cb[(long)b * EMBED_DIM + t];
  s[t] = v * v;
  __syncthreads();
  for (int off = 128; off > 0; off >>= 1) {
    if (t < off) s[t] += s[t + off];
    __syncthreads();
  }
  if (t == 0) { cnorm[b] = s[0]; counts[b] = 0u; }
}

// ---------------------------------------------------------------- main:
// fused distance-GEMM (bf16x3 emulated fp32 via v_wmma_f32_16x16x32_bf16)
// + running argmin + gather/transpose/SSE epilogue. Codebook tiles are
// double-buffered in LDS; next tile is DMA'd with global_load_async_to_lds
// while the current one is consumed by the WMMA pipeline.
__global__ __launch_bounds__(256, 1)
void vq_main(const float* __restrict__ in, const float* __restrict__ cb,
             const float* __restrict__ cnorm, unsigned* __restrict__ counts,
             float* __restrict__ partial, float* __restrict__ out) {
  const int t    = threadIdx.x;
  const int lane = t & 31;
  const int wave = t >> 5;
  const int half = lane >> 4;   // which 16-lane half of the wave
  const int m    = lane & 15;   // A-row / B-col within 16x16 tile
  const int wg   = blockIdx.x;

  const int  n0   = wg * ROWS_PER_WG;     // first flat row (b*HW + hw)
  const int  b    = n0 / HW;              // block never crosses a batch (1024 % 128 == 0)
  const int  hw0  = n0 % HW;
  const long bOff = (long)b * EMBED_DIM * HW;

  __shared__ bf16_t cbh[2][16][CB_PAD];   // bf16-hi tile, ping-pong
  __shared__ bf16_t cbl[2][16][CB_PAD];   // bf16-lo tile, ping-pong
  __shared__ float  cnt16[2][16];
  __shared__ int    rowidx[128];
  __shared__ float  lred[256];
  // 16KB pool: fp32 async staging buffer during the loop, argmin-reduction
  // arrays (128x16 f32 + 128x16 i32 = 16KB) in the epilogue.
  __shared__ __align__(16) char pool[16 * EMBED_DIM * 4];
  float* stage = (float*)pool;

  // ---- build register-resident A fragments: 16 rows x 256 dims, hi/lo bf16.
  // A 16-bit layout: lane m holds row m; half 0 -> K {kb..kb+7, kb+16..kb+23},
  // half 1 -> K {kb+8..kb+15, kb+24..kb+31}.
  v16bf ahi[8], alo[8];
  {
    const int hw_lane = hw0 + wave * 16 + m;
    const float* xb = in + bOff + hw_lane;   // element k at xb[k*HW]
#pragma unroll
    for (int c = 0; c < 8; ++c) {
      const int kb = c * 32;
#pragma unroll
      for (int e = 0; e < 16; ++e) {
        const int k = kb + (e < 8 ? (half * 8 + e) : (16 + half * 8 + (e - 8)));
        float  xv = xb[(long)k * HW];
        bf16_t h  = (bf16_t)xv;
        ahi[c][e] = h;
        alo[c][e] = (bf16_t)(xv - (float)h);
      }
    }
  }

  float best[8];
  int   bidx[8];
#pragma unroll
  for (int j = 0; j < 8; ++j) { best[j] = 3.4e38f; bidx[j] = 0; }

  // ---- tile staging helpers -----------------------------------------------
  // issue: async DMA of the raw fp32 tile (16 codes x 256 dims = 16KB) to LDS
  // convert: fp32 -> bf16 hi/lo split into the ping-pong buffer
#if USE_ASYNC
#define ISSUE_TILE(JT)                                                        \
  {                                                                           \
    const float* g = cb + (long)(JT) * 16 * EMBED_DIM;                        \
    _Pragma("unroll")                                                         \
    for (int rnd = 0; rnd < 4; ++rnd) {                                       \
      const int chunk = rnd * 256 + t;      /* 1024 x 16B chunks */           \
      async_cp16(g + chunk * 4, stage + chunk * 4);                           \
    }                                                                         \
  }
#endif

#define CONVERT_TILE(JT, BUF)                                                 \
  {                                                                           \
    _Pragma("unroll")                                                         \
    for (int j = 0; j < 16; ++j) {                                            \
      float v;                                                                \
      if (USE_ASYNC) v = stage[j * EMBED_DIM + t];                            \
      else           v = cb[(long)(JT) * 16 * EMBED_DIM + j * EMBED_DIM + t]; \
      bf16_t h = (bf16_t)v;                                                   \
      cbh[BUF][j][t] = h;                                                     \
      cbl[BUF][j][t] = (bf16_t)(v - (float)h);                                \
    }                                                                         \
    if (t < 16) cnt16[BUF][t] = cnorm[(JT) * 16 + t];                         \
  }

  // ---- prologue: tile 0 into buffer 0, tile 1 in flight
#if USE_ASYNC
  ISSUE_TILE(0);
  wait_async();
  __syncthreads();
#endif
  CONVERT_TILE(0, 0);
  __syncthreads();
#if USE_ASYNC
  ISSUE_TILE(1);
#else
  __builtin_prefetch(cb + 16 * EMBED_DIM + t * 16, 0, 1);
#endif

  for (int jt = 0; jt < NTILES; ++jt) {
    const int cur = jt & 1;
    const bf16_t (*bhp)[CB_PAD] = cbh[cur];
    const bf16_t (*blp)[CB_PAD] = cbl[cur];

    // ---- 16x16 accumulation over K=256: 8 chunks x 3 WMMAs (hi*hi+lo*hi+hi*lo)
    v8f acc = {};
#pragma unroll
    for (int c = 0; c < 8; ++c) {
      const int k0 = c * 32 + half * 8;
      u16x8 bh0 = *(const u16x8*)&bhp[m][k0];
      u16x8 bh1 = *(const u16x8*)&bhp[m][k0 + 16];
      u16x8 bl0 = *(const u16x8*)&blp[m][k0];
      u16x8 bl1 = *(const u16x8*)&blp[m][k0 + 16];
      v16bf bh = combine8(bh0, bh1);
      v16bf bl = combine8(bl0, bl1);
      acc = __builtin_amdgcn_wmma_f32_16x16x32_bf16(false, ahi[c], false, bh,
                                                    (short)0, acc, false, false);
      acc = __builtin_amdgcn_wmma_f32_16x16x32_bf16(false, alo[c], false, bh,
                                                    (short)0, acc, false, false);
      acc = __builtin_amdgcn_wmma_f32_16x16x32_bf16(false, ahi[c], false, bl,
                                                    (short)0, acc, false, false);
    }

    // ---- fold ||c||^2 - 2 x.c and update running argmin (selects, EXEC intact)
    const float cn   = cnt16[cur][m];
    const int   code = jt * 16 + m;
#pragma unroll
    for (int j = 0; j < 8; ++j) {
      float d = fmaf(-2.0f, acc[j], cn);
      if (d < best[j]) { best[j] = d; bidx[j] = code; }
    }

    // ---- stage tile jt+1 into the other buffer, kick DMA for tile jt+2
    if (jt + 1 < NTILES) {
#if USE_ASYNC
      wait_async();            // tile jt+1 landed in `stage` during compute
#endif
      __syncthreads();         // everyone done with buf[1-cur] & stage ready
      CONVERT_TILE(jt + 1, 1 - cur);
      __syncthreads();         // conversion visible before next compute
#if USE_ASYNC
      if (jt + 2 < NTILES) ISSUE_TILE(jt + 2);
#else
      if (jt + 2 < NTILES)
        __builtin_prefetch(cb + (long)(jt + 2) * 16 * EMBED_DIM + t * 16, 0, 1);
#endif
    }
  }

  // ---- cross-lane argmin: C/D layout -> row = wave*16 + half*8 + j, col = m
  float (*redv)[16] = (float (*)[16])pool;           // stage no longer in use
  int   (*redi)[16] = (int   (*)[16])(pool + 8192);
  __syncthreads();
#pragma unroll
  for (int j = 0; j < 8; ++j) {
    const int row = wave * 16 + half * 8 + j;
    redv[row][m] = best[j];
    redi[row][m] = bidx[j];
  }
  __syncthreads();
  if (t < 128) {
    float bv = redv[t][0];
    int   bi = redi[t][0];
#pragma unroll
    for (int c = 1; c < 16; ++c) {
      float v = redv[t][c];
      int   i = redi[t][c];
      if (v < bv || (v == bv && i < bi)) { bv = v; bi = i; }  // jnp.argmin tie-break
    }
    rowidx[t] = bi;
    out[IDX_OFF + n0 + t] = (float)bi;
    atomicAdd(&counts[bi], 1u);
  }
  __syncthreads();

  // ---- epilogue: quantized_st == codebook[idx] transposed back to (b,c,h,w),
  // plus SSE for the loss. Writes coalesced along hw (128 lanes = 128 rows).
  float lacc = 0.f;
  const int r  = t & 127;
  const int c0 = (t >> 7) * 128;
  const int qi = rowidx[r];
  const float* crow = cb + (long)qi * EMBED_DIM;
  const long   xoff = bOff + hw0 + r;
#pragma unroll 4
  for (int cc = 0; cc < 128; ++cc) {
    const int  c  = c0 + cc;
    float q  = crow[c];
    float xv = in[xoff + (long)c * HW];
    out[Q_OFF + xoff + (long)c * HW] = q;
    float d = q - xv;
    lacc += d * d;
  }
  lred[t] = lacc;
  __syncthreads();
  for (int off = 128; off > 0; off >>= 1) {
    if (t < off) lred[t] += lred[t + off];
    __syncthreads();
  }
  if (t == 0) partial[wg] = lred[0];   // deterministic per-block partial
}

// ---------------------------------------------------------------- finalize:
// loss = 1.6 * mean((q-x)^2); perplexity = exp(-sum p log(p+1e-10))
__global__ void vq_finalize(const unsigned* __restrict__ counts,
                            const float* __restrict__ partial,
                            float* __restrict__ out) {
  __shared__ float s[256];
  const int t = threadIdx.x;
  float acc = 0.f;
  for (int i = t; i < NUM_CODES; i += 256) {
    float p = (float)counts[i] * (1.0f / 32768.0f);
    acc += p * logf(p + 1e-10f);
  }
  s[t] = acc;
  __syncthreads();
  for (int off = 128; off > 0; off >>= 1) {
    if (t < off) s[t] += s[t + off];
    __syncthreads();
  }
  if (t == 0) {
    out[PERP_OFF] = expf(-s[0]);
    float sse = 0.f;
    for (int i = 0; i < N_VEC / ROWS_PER_WG; ++i) sse += partial[i];
    out[0] = 1.6f * sse / (float)Q_SIZE;
  }
}

extern "C" void kernel_launch(void* const* d_in, const int* in_sizes, int n_in,
                              void* d_out, int out_size, void* d_ws, size_t ws_size,
                              hipStream_t stream) {
  (void)in_sizes; (void)n_in; (void)out_size; (void)ws_size;
  const float* in  = (const float*)d_in[0];   // [32, 256, 32, 32] fp32
  const float* cb  = (const float*)d_in[1];   // [8192, 256] fp32
  float*    out    = (float*)d_out;
  unsigned* counts = (unsigned*)d_ws;                       // 8192 u32
  float*    part   = (float*)d_ws + NUM_CODES;              // 256 f32
  float*    cnorm  = (float*)d_ws + NUM_CODES + 256;        // 8192 f32

  vq_prep    <<<NUM_CODES, 256, 0, stream>>>(cb, cnorm, counts);
  vq_main    <<<N_VEC / ROWS_PER_WG, 256, 0, stream>>>(in, cb, cnorm, counts, part, out);
  vq_finalize<<<1, 256, 0, stream>>>(counts, part, out);
}
